// ThreeSystemModel_90220083019993
// MI455X (gfx1250) — matrix operand order repaired and checked by reading, not compiled, not run on hardware
//
#include <hip/hip_runtime.h>
#include <hip/hip_bf16.h>

typedef __attribute__((ext_vector_type(16))) _Float16 v16h;
typedef __attribute__((ext_vector_type(8)))  _Float16 v8h;
typedef __attribute__((ext_vector_type(2)))  _Float16 v2h;
typedef __attribute__((ext_vector_type(8)))  float    v8f;
typedef __attribute__((ext_vector_type(2)))  float    f32x2;
typedef __attribute__((ext_vector_type(4)))  float    f32x4;

#define N_ROWS 32768
#define DIM    512
#define NUM_E  1024
#define NUM_R  256
#define KITERS (DIM / 32)     // 16 WMMA K-steps per dot product
#define CONF_THRESH 0.3f
#define COEFF 0.5f
#define EPS_F 1e-12f

// ---------------------------------------------------------------------------
// Stage 1: L2-normalize one row (f32) and emit f16. One block (256 thr) per
// row of 512; 2 elements per thread.
// ---------------------------------------------------------------------------
__global__ __launch_bounds__(256)
void norm_to_f16(const float* __restrict__ src, _Float16* __restrict__ dst) {
    const int row  = blockIdx.x;
    const int tid  = threadIdx.x;
    const int lane = tid & 31;
    const int wave = tid >> 5;

    const float* p = src + (size_t)row * DIM;
    f32x2 v = *(const f32x2*)(p + tid * 2);
    float ss = v.x * v.x + v.y * v.y;

    #pragma unroll
    for (int off = 16; off > 0; off >>= 1)
        ss += __shfl_xor(ss, off, 32);

    __shared__ float wsum[8];
    if (lane == 0) wsum[wave] = ss;
    __syncthreads();

    float total = 0.f;
    #pragma unroll
    for (int i = 0; i < 8; ++i) total += wsum[i];

    const float rn = 1.0f / fmaxf(sqrtf(total), EPS_F);
    v2h o;
    o.x = (_Float16)(v.x * rn);
    o.y = (_Float16)(v.y * rn);
    *(v2h*)(dst + (size_t)row * DIM + tid * 2) = o;
}

// ---------------------------------------------------------------------------
// Stage 2: fused similarity GEMM + row argmax, 2 row-stripes per wave.
//
// WMMA mapping: A = centroid tile (M=16 centroids x K=32), B = residual tile
// (K=32 x N=16 rows). D[M=centroid][N=row]; lane L holds row N=L&15 and 8
// centroid results (M = r + (L>=16 ? 8 : 0)).
//
// Per-lane operand layouts used here (contiguous k, not the ISA table order; 16-bit, wave32):
//   A: lane holds M=lane&15; K = a_off+{0..15}, a_off = (lane<16 ? 0 : 16)
//      -> one contiguous 32B v16h load
//   B: lane holds N=lane&15; K = b_off+{0..15}, b_off = (lane<16 ? 0 : 16)
//      -> one contiguous 32B v16h load
//
// Each wave owns 32 residual rows (two stripes), keeps BOTH full-D B operands
// resident in VGPRs (2 x 128 VGPRs), and issues two independent WMMAs per
// A-fragment load: halves centroid traffic, doubles FLOP/byte, and provides
// two accumulator chains to hide the A-load latency.
// ---------------------------------------------------------------------------
__device__ __forceinline__ v16h load_a_frag(const _Float16* row, int k, int a_off) {
    return *(const v16h*)(row + k + a_off);
}

__device__ __forceinline__ void argmax_tiles2(const _Float16* __restrict__ centn,
                                              const v16h (&B0)[KITERS],
                                              const v16h (&B1)[KITERS],
                                              int num_cent, int nlo, int a_off,
                                              int id_half,
                                              float& best0, int& bid0,
                                              float& best1, int& bid1) {
    for (int cb = 0; cb < num_cent; cb += 16) {
        const _Float16* centRow = centn + (size_t)(cb + nlo) * DIM;
        v8f acc0 = {};
        v8f acc1 = {};
        #pragma unroll
        for (int i = 0; i < KITERS; ++i) {
            v16h A = load_a_frag(centRow, i * 32, a_off);
            acc0 = __builtin_amdgcn_wmma_f32_16x16x32_f16(
                false, A, false, B0[i], (short)0, acc0, false, false);
            acc1 = __builtin_amdgcn_wmma_f32_16x16x32_f16(
                false, A, false, B1[i], (short)0, acc1, false, false);
        }
        const int idbase = cb + id_half;
        #pragma unroll
        for (int r = 0; r < 8; ++r) {
            float s0 = acc0[r];
            if (s0 > best0) { best0 = s0; bid0 = idbase + r; }
            float s1 = acc1[r];
            if (s1 > best1) { best1 = s1; bid1 = idbase + r; }
        }
    }
    // merge the two half-rows; prefer lower id on tie (jnp.argmax semantics)
    float ob0 = __shfl_xor(best0, 16, 32);
    int   oi0 = __shfl_xor(bid0,  16, 32);
    if (ob0 > best0 || (ob0 == best0 && oi0 < bid0)) { best0 = ob0; bid0 = oi0; }
    float ob1 = __shfl_xor(best1, 16, 32);
    int   oi1 = __shfl_xor(bid1,  16, 32);
    if (ob1 > best1 || (ob1 == best1 && oi1 < bid1)) { best1 = ob1; bid1 = oi1; }
}

__global__ __launch_bounds__(256)
void sim_argmax(const _Float16* __restrict__ resn,
                const _Float16* __restrict__ entn,
                const _Float16* __restrict__ reln,
                const float*    __restrict__ lookup_mask,
                int*            __restrict__ pair_idx,
                float*          __restrict__ scale_out) {
    const int lane = threadIdx.x & 31;
    const int wave = threadIdx.x >> 5;
    const int row_base = (blockIdx.x * 8 + wave) * 32;   // 32 rows per wave

    const int  nlo     = lane & 15;
    const bool hiHalf  = lane >= 16;
    const int  a_off   = hiHalf ? 16 : 0;
    const int  b_off   = hiHalf ? 16 : 0;
    const int  id_half = hiHalf ? 8 : 0;

    const _Float16* resRow0 = resn + (size_t)(row_base + nlo) * DIM;
    const _Float16* resRow1 = resn + (size_t)(row_base + 16 + nlo) * DIM;

    // Preload both stripes' B operands for the full K range (2 x 128 VGPRs).
    v16h B0[KITERS], B1[KITERS];
    #pragma unroll
    for (int i = 0; i < KITERS; ++i) {
        B0[i] = *(const v16h*)(resRow0 + i * 32 + b_off);
        B1[i] = *(const v16h*)(resRow1 + i * 32 + b_off);
    }

    float e_best0 = -1e30f, e_best1 = -1e30f;
    int   e_bid0 = 0, e_bid1 = 0;
    argmax_tiles2(entn, B0, B1, NUM_E, nlo, a_off, id_half,
                  e_best0, e_bid0, e_best1, e_bid1);

    float r_best0 = -1e30f, r_best1 = -1e30f;
    int   r_bid0 = 0, r_bid1 = 0;
    argmax_tiles2(reln, B0, B1, NUM_R, nlo, a_off, id_half,
                  r_best0, r_bid0, r_best1, r_bid1);

    if (lane < 16) {
        {
            const int row  = row_base + lane;
            const int pair = e_bid0 * NUM_R + r_bid0;
            const float m  = lookup_mask[(size_t)pair];
            const float ok = (e_best0 > CONF_THRESH && r_best0 > CONF_THRESH) ? 1.0f : 0.0f;
            pair_idx[row]  = pair;
            scale_out[row] = m * ok * COEFF;
        }
        {
            const int row  = row_base + 16 + lane;
            const int pair = e_bid1 * NUM_R + r_bid1;
            const float m  = lookup_mask[(size_t)pair];
            const float ok = (e_best1 > CONF_THRESH && r_best1 > CONF_THRESH) ? 1.0f : 0.0f;
            pair_idx[row]  = pair;
            scale_out[row] = m * ok * COEFF;
        }
    }
}

// ---------------------------------------------------------------------------
// Stage 3: gather lookup_table row, scale, store. 1 block per output row,
// 128 threads x float4 = 512 floats. Pure HBM-bandwidth bound.
// ---------------------------------------------------------------------------
__global__ __launch_bounds__(128)
void gather_scale(const float* __restrict__ table,
                  const int*   __restrict__ pair_idx,
                  const float* __restrict__ scale,
                  float*       __restrict__ out) {
    const int row = blockIdx.x;
    const int t   = threadIdx.x;
    const float sc = scale[row];
    const f32x4* src = (const f32x4*)(table + (size_t)pair_idx[row] * DIM);
    f32x4 v = src[t];
    v.x *= sc; v.y *= sc; v.z *= sc; v.w *= sc;
    ((f32x4*)(out + (size_t)row * DIM))[t] = v;
}

// ---------------------------------------------------------------------------
extern "C" void kernel_launch(void* const* d_in, const int* in_sizes, int n_in,
                              void* d_out, int out_size, void* d_ws, size_t ws_size,
                              hipStream_t stream) {
    const float* residuals = (const float*)d_in[0];   // (N, D)
    const float* ent_cent  = (const float*)d_in[1];   // (NE, D)
    const float* rel_cent  = (const float*)d_in[2];   // (NR, D)
    const float* table     = (const float*)d_in[3];   // (NE, NR, D)
    const float* mask      = (const float*)d_in[4];   // (NE, NR)
    float* out = (float*)d_out;

    // workspace layout (all offsets multiples of 256B)
    char* ws = (char*)d_ws;
    size_t off = 0;
    _Float16* resn = (_Float16*)(ws + off); off += (size_t)N_ROWS * DIM * sizeof(_Float16); // 32 MB
    _Float16* entn = (_Float16*)(ws + off); off += (size_t)NUM_E  * DIM * sizeof(_Float16); // 1 MB
    _Float16* reln = (_Float16*)(ws + off); off += (size_t)NUM_R  * DIM * sizeof(_Float16); // 256 KB
    int*   pidx = (int*)(ws + off);   off += (size_t)N_ROWS * sizeof(int);                  // 128 KB
    float* scl  = (float*)(ws + off);                                                       // 128 KB

    norm_to_f16<<<N_ROWS, 256, 0, stream>>>(residuals, resn);
    norm_to_f16<<<NUM_E,  256, 0, stream>>>(ent_cent,  entn);
    norm_to_f16<<<NUM_R,  256, 0, stream>>>(rel_cent,  reln);

    // 8 waves/block * 32 rows/wave = 256 rows per block
    sim_argmax<<<N_ROWS / 256, 256, 0, stream>>>(resn, entn, reln, mask, pidx, scl);

    gather_scale<<<N_ROWS, 128, 0, stream>>>(table, pidx, scl, out);
}
